// LstmRlModel_10995116278119
// MI455X (gfx1250) — matrix operand order repaired and checked by reading, not compiled, not run on hardware
//
#include <hip/hip_runtime.h>
#include <hip/hip_bf16.h>
#include <math.h>

// ---------------------------------------------------------------------------
// Problem constants (from reference)
// ---------------------------------------------------------------------------
#define BB   4096   // batch
#define NN   64     // agents / time steps
#define DD   12     // features per agent
#define SSZ  5      // self-state size
#define II   7      // LSTM input size (D - SS)
#define HH   256    // LSTM hidden
#define G4H  1024   // 4*H gates
#define KC   288    // concatenated K for LSTM WMMA: 256 (h) + 32 (x pad)
#define KCP  296    // LDS row stride (pad: 592B = 148 dw, bank-conflict free)
#define M1N  512
#define M2N  512
#define AN   81
#define ANP  96     // A padded to 6 16-col tiles
#define K1P  288    // 261 padded to 9 K-tiles of 32
#define LROWS 32    // batch rows per LSTM block (two 16-row tiles per wave)

typedef __attribute__((ext_vector_type(16))) __bf16 v16bf;
typedef __attribute__((ext_vector_type(8)))  __bf16 v8bf;
typedef __attribute__((ext_vector_type(8)))  float  v8f;

// Explicit global-address-space pointers: keeps laundered pointers on the
// global_load path (LOADcnt only) instead of flat_load (LOADcnt+DScnt).
typedef const __bf16 __attribute__((address_space(1)))* gcbf16p;
typedef const v16bf  __attribute__((address_space(1)))* gcv16p;

#define WMMA_BF16(a, b, c) \
  __builtin_amdgcn_wmma_f32_16x16x32_bf16(false, (a), false, (b), (short)0, (c), false, false)

static __device__ __forceinline__ float sigf(float x) {
  return 1.0f / (1.0f + __expf(-x));
}

// A-operand fragment (16x32 bf16, MxK), contiguous K order matching B:
//   lanes 0-15  : row m = lane,    elements hold K=0..15
//   lanes 16-31 : row m = lane-16, elements hold K=16..31
// caller passes rowp = &A[row][0], k = ktile*32 + (lane<16 ? 0 : 16)
static __device__ __forceinline__ v16bf load_afrag(const __bf16* rowp, int k) {
  v8bf lo = *(const v8bf*)(rowp + k);
  v8bf hi = *(const v8bf*)(rowp + k + 8);
  v16bf r;
#pragma unroll
  for (int i = 0; i < 8; ++i) { r[i] = lo[i]; r[i + 8] = hi[i]; }
  return r;
}

// ---------------------------------------------------------------------------
// Kernel 1: build Wcat[1024][288] bf16 = [W_hh | W_ih | 0], bias_sum[1024]
// ---------------------------------------------------------------------------
__global__ void k_prep_wcat(const float* __restrict__ W_ih,   // [1024][7]
                            const float* __restrict__ W_hh,   // [1024][256]
                            const float* __restrict__ b_ih,
                            const float* __restrict__ b_hh,
                            __bf16* __restrict__ Wcat,        // [1024][288]
                            float* __restrict__ bias_sum) {   // [1024]
  int n = blockIdx.x, t = threadIdx.x;
  Wcat[n * KC + t] = (__bf16)W_hh[n * HH + t];
  if (t < 32) {
    float v = (t < II) ? W_ih[n * II + t] : 0.0f;
    Wcat[n * KC + HH + t] = (__bf16)v;
  }
  if (t == 0) bias_sum[n] = b_ih[n] + b_hh[n];
}

// ---------------------------------------------------------------------------
// Kernel 2: transpose f32 [K][N] -> bf16 [Npad][Kpad], zero-padded.
// grid = Npad blocks, block = Kpad threads.
// ---------------------------------------------------------------------------
__global__ void k_transpose(const float* __restrict__ in, int K, int N,
                            int Kpad, __bf16* __restrict__ out) {
  int n = blockIdx.x, k = threadIdx.x;
  float v = (n < N && k < K) ? in[(size_t)k * N + n] : 0.0f;
  out[(size_t)n * Kpad + k] = (__bf16)v;
}

// ---------------------------------------------------------------------------
// Kernel 3: per-row distance sort (bitonic, 64 elems) + gather features
// grid = 4096, block = 64
// ---------------------------------------------------------------------------
__global__ void k_sort_gather(const float* __restrict__ state,   // [B][64][12]
                              __bf16* __restrict__ xbf,          // [B][64][8]
                              __bf16* __restrict__ joint) {      // [B][288]
  int b = blockIdx.x, t = threadIdx.x;
  __shared__ float key[NN];
  __shared__ int   sidx[NN];
  const float* srow = state + (size_t)b * NN * DD;

  float s5 = srow[t * DD + 5], s6 = srow[t * DD + 6];
  float d  = (s5 != 0.0f && s6 != 0.0f) ? sqrtf(s5 * s5 + s6 * s6)
                                        : __builtin_inff();
  key[t]  = -d;  // ascending sort of -d == descending sort of d
  sidx[t] = t;
  __syncthreads();

  for (int k = 2; k <= NN; k <<= 1) {
    for (int j = k >> 1; j > 0; j >>= 1) {
      int ixj = t ^ j;
      if (ixj > t) {
        bool up = ((t & k) == 0);
        float ki = key[t], kj = key[ixj];
        bool sw = up ? (ki > kj) : (ki < kj);
        if (sw) {
          key[t] = kj; key[ixj] = ki;
          int tmp = sidx[t]; sidx[t] = sidx[ixj]; sidx[ixj] = tmp;
        }
      }
      __syncthreads();
    }
  }

  int src = sidx[t];
#pragma unroll
  for (int kk = 0; kk < II; ++kk)
    xbf[((size_t)b * NN + t) * 8 + kk] = (__bf16)srow[src * DD + SSZ + kk];
  xbf[((size_t)b * NN + t) * 8 + II] = (__bf16)0.0f;

  if (t < SSZ) joint[(size_t)b * KC + t] = (__bf16)srow[t];      // state[b][0][0:5]
  if (t >= 32 && t < 32 + (K1P - (SSZ + HH)))                    // zero pad 261..287
    joint[(size_t)b * KC + (SSZ + HH) + (t - 32)] = (__bf16)0.0f;
}

// ---------------------------------------------------------------------------
// Kernel 4: LSTM recurrence. Block = 32 batch rows, 256 threads (8 waves).
// Wave w owns hidden cols [32w, 32w+32) across ALL four gates (so cell state c
// never leaves registers) and TWO 16-row tiles (each B fragment feeds 2 WMMAs,
// halving L2 weight traffic). h is double-buffered in LDS as the WMMA A
// operand [h | x_t]. The Wcat pointer is laundered through inline asm each
// time step (defeats loop-invariant hoisting -> scratch spilling) and cast to
// address_space(1) so the weight stream stays on global_load_b128/LOADcnt,
// decoupled from the LDS A-fragment DScnt waits.
// ---------------------------------------------------------------------------
__global__ void __launch_bounds__(256, 1)
k_lstm(const __bf16* __restrict__ xbf,    // [B][64][8]
       const __bf16* __restrict__ Wcat,   // [1024][288]
       const float*  __restrict__ bias,   // [1024]
       __bf16* __restrict__ joint) {      // [B][288] (writes cols 5..260)
  __shared__ __bf16 hx[2][LROWS][KCP];    // ~37 KB

  const int tid  = threadIdx.x;
  const int lane = tid & 31;
  const int w    = tid >> 5;              // wave 0..7
  const int wcol0 = w * 32;               // hidden columns owned by this wave
  const int b0   = blockIdx.x * LROWS;
  const int l15  = lane & 15;
  const int hiA  = (lane & 16) ? 16 : 0;  // A-frag K offset
  const int hiB  = (lane & 16) ? 16 : 0;  // B-frag K offset
  const int mAdd = (lane & 16) ? 8 : 0;   // C/D row offset for this lane half

  // init LDS: zero both buffers, then stage x_0
  for (int i = tid; i < 2 * LROWS * KCP; i += 256)
    ((__bf16*)hx)[i] = (__bf16)0.0f;
  __syncthreads();
  {
    int m = tid >> 3, k = tid & 7;        // 256 threads = 32 rows x 8
    hx[0][m][HH + k] = xbf[((size_t)(b0 + m) * NN + 0) * 8 + k];
  }

  float c[2][16];
#pragma unroll
  for (int rt = 0; rt < 2; ++rt)
#pragma unroll
    for (int i = 0; i < 16; ++i) c[rt][i] = 0.0f;

  // per-tile bias splat: tile index = gate*2 + colhalf
  float bs[8];
#pragma unroll
  for (int g = 0; g < 4; ++g)
#pragma unroll
    for (int ch = 0; ch < 2; ++ch)
      bs[g * 2 + ch] = bias[g * HH + wcol0 + ch * 16 + l15];

  unsigned long long wlaund = (unsigned long long)Wcat;
  __syncthreads();

  for (int t = 0; t < NN; ++t) {
    const int cur = t & 1, nxt = cur ^ 1;

    // Launder the weight pointer: defeats loop-invariant hoisting (which
    // would need 576 VGPRs of B fragments and spill to scratch).
    asm volatile("" : "+s"(wlaund));
    gcbf16p wptr = (gcbf16p)wlaund;       // explicit global address space

    v8f acc[2][8];
#pragma unroll
    for (int rt = 0; rt < 2; ++rt)
#pragma unroll
      for (int i = 0; i < 8; ++i)
#pragma unroll
        for (int r = 0; r < 8; ++r) acc[rt][i][r] = bs[i];

    const __bf16* arow0 = &hx[cur][l15][0];
    const __bf16* arow1 = &hx[cur][16 + l15][0];
#pragma unroll
    for (int kt = 0; kt < KC / 32; ++kt) {
      const int kb = kt * 32;
      v16bf a0 = load_afrag(arow0, kb + hiA);
      v16bf a1 = load_afrag(arow1, kb + hiA);
#pragma unroll
      for (int g = 0; g < 4; ++g)
#pragma unroll
        for (int ch = 0; ch < 2; ++ch) {
          const int n = g * HH + wcol0 + ch * 16 + l15;
          v16bf bm = *(gcv16p)(wptr + (size_t)n * KC + kb + hiB);
          acc[0][g * 2 + ch] = WMMA_BF16(a0, bm, acc[0][g * 2 + ch]);
          acc[1][g * 2 + ch] = WMMA_BF16(a1, bm, acc[1][g * 2 + ch]);
        }
    }

    // gate nonlinearities; c in registers, h -> LDS (next buffer)
#pragma unroll
    for (int rt = 0; rt < 2; ++rt)
#pragma unroll
      for (int ch = 0; ch < 2; ++ch)
#pragma unroll
        for (int r = 0; r < 8; ++r) {
          float gi = acc[rt][0 * 2 + ch][r];
          float gf = acc[rt][1 * 2 + ch][r];
          float gg = acc[rt][2 * 2 + ch][r];
          float go = acc[rt][3 * 2 + ch][r];
          float cc = c[rt][ch * 8 + r];
          cc = sigf(gf) * cc + sigf(gi) * tanhf(gg);
          c[rt][ch * 8 + r] = cc;
          float h = sigf(go) * tanhf(cc);
          const int m = rt * 16 + r + mAdd;     // row within 32-row block
          const int n = wcol0 + ch * 16 + l15;  // hidden column
          hx[nxt][m][n] = (__bf16)h;
          if (t == NN - 1)
            joint[(size_t)(b0 + m) * KC + SSZ + n] = (__bf16)h;
        }

    // stage x_{t+1}: 256 threads = 32 rows x 8 cols
    if (t < NN - 1) {
      int m = tid >> 3, k = tid & 7;
      hx[nxt][m][HH + k] = xbf[((size_t)(b0 + m) * NN + (t + 1)) * 8 + k];
    }
    __syncthreads();
  }
}

// ---------------------------------------------------------------------------
// Kernel 5: generic WMMA GEMM  out = act(A[M][Kpad] @ Bt[Npad][Kpad]^T + bias)
// block 256 threads = 8 waves -> 64x32 tile; grid (M/64, Npad/32)
// ---------------------------------------------------------------------------
__global__ void __launch_bounds__(256, 2)
k_gemm(const __bf16* __restrict__ A, int lda,
       const __bf16* __restrict__ Bt, int ldb,
       const float* __restrict__ bias,
       __bf16* __restrict__ outB, float* __restrict__ outF, int ldo,
       int Nact, int Kpad, int doRelu) {
  const int tid  = threadIdx.x;
  const int lane = tid & 31;
  const int w    = tid >> 5;
  const int m0   = blockIdx.x * 64 + (w & 3) * 16;
  const int n0   = blockIdx.y * 32 + (w >> 2) * 16;
  const int l15  = lane & 15;
  const int hiA  = (lane & 16) ? 16 : 0;
  const int hiB  = (lane & 16) ? 16 : 0;

  const __bf16* arow = A + (size_t)(m0 + l15) * lda;
  const __bf16* brow = Bt + (size_t)(n0 + l15) * ldb;

  v8f acc;
#pragma unroll
  for (int r = 0; r < 8; ++r) acc[r] = 0.0f;

  for (int kb = 0; kb < Kpad; kb += 32) {
    v16bf a  = load_afrag(arow, kb + hiA);
    v16bf bm = *(const v16bf*)(brow + kb + hiB);
    acc = WMMA_BF16(a, bm, acc);
  }

  const int n = n0 + l15;
  if (n < Nact) {
    const float bv = bias[n];
#pragma unroll
    for (int r = 0; r < 8; ++r) {
      const int m = m0 + r + ((lane & 16) ? 8 : 0);
      float v = acc[r] + bv;
      if (doRelu) v = v > 0.0f ? v : 0.0f;
      if (outB) outB[(size_t)m * ldo + n] = (__bf16)v;
      else      outF[(size_t)m * ldo + n] = v;
    }
  }
}

// ---------------------------------------------------------------------------
// Host launcher
// ---------------------------------------------------------------------------
extern "C" void kernel_launch(void* const* d_in, const int* in_sizes, int n_in,
                              void* d_out, int out_size, void* d_ws, size_t ws_size,
                              hipStream_t stream) {
  const float* state = (const float*)d_in[0];
  const float* W_ih  = (const float*)d_in[1];
  const float* W_hh  = (const float*)d_in[2];
  const float* b_ih  = (const float*)d_in[3];
  const float* b_hh  = (const float*)d_in[4];
  const float* W1    = (const float*)d_in[5];
  const float* b1    = (const float*)d_in[6];
  const float* W2    = (const float*)d_in[7];
  const float* b2    = (const float*)d_in[8];
  const float* Wv    = (const float*)d_in[9];
  const float* bv    = (const float*)d_in[10];
  float* out = (float*)d_out;

  // workspace carve-out (256B aligned)
  char* ws = (char*)d_ws;
  size_t off = 0;
  auto alloc = [&](size_t bytes) -> void* {
    off = (off + 255) & ~(size_t)255;
    void* p = ws + off;
    off += bytes;
    return p;
  };
  __bf16* xbf      = (__bf16*)alloc((size_t)BB * NN * 8 * 2);     // 4 MB
  __bf16* joint    = (__bf16*)alloc((size_t)BB * KC * 2);         // 2.25 MB
  __bf16* out1     = (__bf16*)alloc((size_t)BB * M1N * 2);        // 4 MB
  __bf16* out2     = (__bf16*)alloc((size_t)BB * M2N * 2);        // 4 MB
  __bf16* Wcat     = (__bf16*)alloc((size_t)G4H * KC * 2);
  float*  bias_sum = (float*)alloc((size_t)G4H * 4);
  __bf16* W1t      = (__bf16*)alloc((size_t)M1N * K1P * 2);
  __bf16* W2t      = (__bf16*)alloc((size_t)M2N * M1N * 2);
  __bf16* Wvt      = (__bf16*)alloc((size_t)ANP * M2N * 2);
  (void)ws_size; (void)n_in; (void)in_sizes; (void)out_size;

  // weight prep
  k_prep_wcat<<<G4H, 256, 0, stream>>>(W_ih, W_hh, b_ih, b_hh, Wcat, bias_sum);
  k_transpose<<<M1N, K1P, 0, stream>>>(W1, SSZ + HH, M1N, K1P, W1t);   // 261x512 -> [512][288]
  k_transpose<<<M2N, M1N, 0, stream>>>(W2, M1N, M2N, M1N, W2t);        // 512x512 -> [512][512]
  k_transpose<<<ANP, M2N, 0, stream>>>(Wv, M2N, AN, M2N, Wvt);         // 512x81  -> [96][512]

  // sort + gather + joint prep
  k_sort_gather<<<BB, NN, 0, stream>>>(state, xbf, joint);

  // LSTM recurrence
  k_lstm<<<BB / LROWS, 256, 0, stream>>>(xbf, Wcat, bias_sum, joint);

  // MLP head
  dim3 g1(BB / 64, M1N / 32);
  k_gemm<<<g1, 256, 0, stream>>>(joint, KC, W1t, K1P, b1, out1, nullptr, M1N,
                                 M1N, K1P, 1);
  dim3 g2(BB / 64, M2N / 32);
  k_gemm<<<g2, 256, 0, stream>>>(out1, M1N, W2t, M1N, b2, out2, nullptr, M2N,
                                 M2N, M1N, 1);
  dim3 g3(BB / 64, ANP / 32);
  k_gemm<<<g3, 256, 0, stream>>>(out2, M2N, Wvt, M2N, bv, nullptr, out, AN,
                                 AN, M2N, 0);
}
